// ConceptEmbedding_20693152432345
// MI455X (gfx1250) — matrix operand order repaired and checked by reading, not compiled, not run on hardware
//
#include <hip/hip_runtime.h>

typedef __attribute__((ext_vector_type(16))) __bf16 v16bf;
typedef __attribute__((ext_vector_type(8)))  __bf16 v8bf;
typedef __attribute__((ext_vector_type(8)))  float  v8f;
typedef __attribute__((ext_vector_type(4)))  float  v4f;

#define IN_FEATURES 2048
#define N_CONCEPTS  64
#define EMB_SIZE    128
#define BATCH       8192
#define NEG_SLOPE   0.01f

static __device__ __forceinline__ float leaky(float v) {
    return v >= 0.f ? v : NEG_SLOPE * v;
}

// Issue one per-lane async 16B copy global -> LDS. IOFFSET is added to BOTH
// the LDS and the global address (ISA 15.18.3), so one base pair serves a
// whole strided sequence of chunks.
#define ASYNC_B128(ldsoff, gaddr, OFF)                                        \
    asm volatile("global_load_async_to_lds_b128 %0, %1, off offset:%c2"       \
                 :: "v"(ldsoff), "v"(gaddr), "n"(OFF) : "memory")

static __device__ __forceinline__ void wait_async_all() {
    asm volatile("s_wait_asynccnt 0x0" ::: "memory");
}

// ---------------------------------------------------------------------------
// Kernel 0: convert We (128x2048) and Wc (64x256x128) f32 -> bf16 scratch.
// ---------------------------------------------------------------------------
__global__ __launch_bounds__(256) void cvt_weights(
    const float* __restrict__ We, const float* __restrict__ Wc,
    __bf16* __restrict__ we_b, __bf16* __restrict__ wc_b) {
    const int stride = gridDim.x * blockDim.x;
    int i = blockIdx.x * blockDim.x + threadIdx.x;
    for (int idx = i; idx < EMB_SIZE * IN_FEATURES; idx += stride)
        we_b[idx] = (__bf16)We[idx];
    for (int idx = i; idx < N_CONCEPTS * 2 * EMB_SIZE * EMB_SIZE; idx += stride)
        wc_b[idx] = (__bf16)Wc[idx];
}

// ---------------------------------------------------------------------------
// Kernel 1: emb = leaky_relu(x @ We.T + be)
//   M=8192 N=128 K=2048.  Block = 256 thr (8 waves) owns 128 batch rows.
//   We-bf16 K-slices (128K x 128N = 32 KB) are async double-buffered in LDS
//   and shared by all 8 waves; x streams from HBM with inline f32->bf16
//   conversion (single-use data).
// ---------------------------------------------------------------------------
__global__ __launch_bounds__(256) void gemm1_emb(
    const float* __restrict__ x, const __bf16* __restrict__ we_b,
    const float* __restrict__ be,
    float* __restrict__ emb_out, __bf16* __restrict__ emb_b) {

    extern __shared__ __bf16 smem[];
    __bf16* lws0 = smem;                    // [row 0..127][k 0..127] 32 KB
    __bf16* lws1 = smem + 128 * 128;

    const int tid  = threadIdx.x;
    const int lane = tid & 31;
    const int wave = tid >> 5;
    const int n16  = lane & 15;
    const int hi   = lane >> 4;
    const int b0   = blockIdx.x * 128;
    const int row0 = wave * 16;

    // stage We[:, kc:kc+128] -> LDS[row*128 + kcol]; thread covers row tid/2,
    // k-half tid%2 => 128 contiguous bytes on both sides.
    auto stage_we = [&](__bf16* buf, int kc) {
        unsigned loff = (unsigned)(size_t)buf + tid * 128;
        const char* ga = (const char*)(we_b +
            (size_t)(tid >> 1) * IN_FEATURES + kc + (tid & 1) * 64);
#pragma unroll
        for (int c = 0; c < 8; ++c) ASYNC_B128(loff, ga, c * 16);
    };

    v8f acc[8];
#pragma unroll
    for (int t = 0; t < 8; ++t) acc[t] = (v8f){};

    const float* xrow = x + (size_t)(b0 + row0 + n16) * IN_FEATURES;

    stage_we(lws0, 0);
    wait_async_all();
    __syncthreads();

    for (int kc = 0; kc < IN_FEATURES; kc += 128) {
        const __bf16* wl = ((kc >> 7) & 1) ? lws1 : lws0;
        if (kc + 128 < IN_FEATURES)
            stage_we(((kc >> 7) & 1) ? lws0 : lws1, kc + 128);

#pragma unroll
        for (int kb = 0; kb < 128; kb += 32) {
            // A fragment: two contiguous 8-elem chunks at K = 16*hi, 16*hi+8
            v16bf a;
            {
                const float* p1 = xrow + kc + kb + 16 * hi;
                const float* p2 = xrow + kc + kb + 16 * hi + 8;
                v4f f0 = *(const v4f*)(p1);
                v4f f1 = *(const v4f*)(p1 + 4);
                v4f f2 = *(const v4f*)(p2);
                v4f f3 = *(const v4f*)(p2 + 4);
#pragma unroll
                for (int i = 0; i < 4; ++i) {
                    a[i]      = (__bf16)f0[i];
                    a[4 + i]  = (__bf16)f1[i];
                    a[8 + i]  = (__bf16)f2[i];
                    a[12 + i] = (__bf16)f3[i];
                }
            }
#pragma unroll
            for (int t = 0; t < 8; ++t) {
                v16bf b = *(const v16bf*)(wl +
                          (16 * t + n16) * 128 + kb + 16 * hi);
                acc[t] = __builtin_amdgcn_wmma_f32_16x16x32_bf16(
                    false, a, false, b, (short)0, acc[t], false, false);
            }
        }
        wait_async_all();
        __syncthreads();
    }

    // epilogue: bias + leaky relu, store f32 emb + bf16 emb
#pragma unroll
    for (int t = 0; t < 8; ++t) {
        const int n = 16 * t + n16;
        const float bias = be[n];
#pragma unroll
        for (int j = 0; j < 8; ++j) {
            const int m = row0 + j + 8 * hi;
            float v = leaky(acc[t][j] + bias);
            size_t off = (size_t)(b0 + m) * EMB_SIZE + n;
            emb_out[off] = v;
            emb_b[off]   = (__bf16)v;
        }
    }
}

// ---------------------------------------------------------------------------
// Kernel 2 (fused): context = leaky(emb @ Wc[k].T + bc[k]);
//                   g = sigmoid(context . wp + bp);  c_emb = pos*g + neg*(1-g)
//   Block = 256 thr (8 waves) owns 128 batch rows; all waves share concept k.
//   emb tile (32 KB) async-staged once; Wc[k] (64 KB) async double-buffered
//   in LDS (160 KB dynamic LDS) so concept k+1 streams in from L2 via the
//   ASYNCcnt path while concept k runs WMMAs out of LDS.
// ---------------------------------------------------------------------------
__global__ __launch_bounds__(256) void gemm2_fused(
    const __bf16* __restrict__ emb_b, const __bf16* __restrict__ wc_b,
    const float* __restrict__ bc, const float* __restrict__ wp,
    const float* __restrict__ bp,
    float* __restrict__ c_emb, float* __restrict__ c_pred,
    float* __restrict__ c_int) {

    extern __shared__ __bf16 smem[];
    __bf16* lds_emb = smem;                       // 128*128 bf16 = 32 KB
    __bf16* lds_wc0 = smem + 128 * EMB_SIZE;      // 64 KB
    __bf16* lds_wc1 = lds_wc0 + 2 * EMB_SIZE * EMB_SIZE;

    const int tid  = threadIdx.x;
    const int lane = tid & 31;
    const int wave = tid >> 5;
    const int n16  = lane & 15;
    const int hi   = lane >> 4;
    const int b0   = blockIdx.x * 128;
    const int row0 = wave * 16;

    // ---- prologue: async-stage emb tile (128B/thr) + Wc[0] (256B/thr) ----
    {
        unsigned loff = (unsigned)(size_t)lds_emb + tid * 128;
        const char* ga = (const char*)(emb_b + (size_t)b0 * EMB_SIZE) + tid * 128;
#pragma unroll
        for (int c = 0; c < 8; ++c) ASYNC_B128(loff, ga, c * 16);
    }
    {
        unsigned loff = (unsigned)(size_t)lds_wc0 + tid * 256;
        const char* ga = (const char*)wc_b + tid * 256;
#pragma unroll
        for (int c = 0; c < 16; ++c) ASYNC_B128(loff, ga, c * 16);
    }
    wait_async_all();
    __syncthreads();

    float wpv[16];
#pragma unroll
    for (int t = 0; t < 16; ++t) wpv[t] = wp[16 * t + n16];
    const float bp0 = bp[0];

    for (int k = 0; k < N_CONCEPTS; ++k) {
        const __bf16* wl = (k & 1) ? lds_wc1 : lds_wc0;

        // kick off async stage of concept k+1 into the other buffer
        if (k + 1 < N_CONCEPTS) {
            __bf16* nb = (k & 1) ? lds_wc0 : lds_wc1;
            unsigned loff = (unsigned)(size_t)nb + tid * 256;
            const char* ga = (const char*)(wc_b +
                (size_t)(k + 1) * 2 * EMB_SIZE * EMB_SIZE) + tid * 256;
#pragma unroll
            for (int c = 0; c < 16; ++c) ASYNC_B128(loff, ga, c * 16);
        }

        v8f acc[16];
#pragma unroll
        for (int t = 0; t < 16; ++t) acc[t] = (v8f){};

#pragma unroll
        for (int kb = 0; kb < EMB_SIZE; kb += 32) {
            v8bf c1 = *(const v8bf*)(lds_emb +
                       (row0 + n16) * EMB_SIZE + kb + 16 * hi);
            v8bf c2 = *(const v8bf*)(lds_emb +
                       (row0 + n16) * EMB_SIZE + kb + 16 * hi + 8);
            v16bf a = __builtin_shufflevector(
                c1, c2, 0, 1, 2, 3, 4, 5, 6, 7, 8, 9, 10, 11, 12, 13, 14, 15);
#pragma unroll
            for (int t = 0; t < 16; ++t) {
                v16bf b = *(const v16bf*)(wl +
                          (size_t)(16 * t + n16) * EMB_SIZE + kb + 16 * hi);
                acc[t] = __builtin_amdgcn_wmma_f32_16x16x32_bf16(
                    false, a, false, b, (short)0, acc[t], false, false);
            }
        }

        // bias + leaky relu + partial dot with wp
        float p[8];
#pragma unroll
        for (int j = 0; j < 8; ++j) p[j] = 0.f;
#pragma unroll
        for (int t = 0; t < 16; ++t) {
            const float bias = bc[k * 2 * EMB_SIZE + 16 * t + n16];
#pragma unroll
            for (int j = 0; j < 8; ++j) {
                float v = leaky(acc[t][j] + bias);
                acc[t][j] = v;
                p[j] += v * wpv[t];
            }
        }

        // reduce across the 16-lane N-group, sigmoid, gate, store
#pragma unroll
        for (int j = 0; j < 8; ++j) {
            float s = p[j];
#pragma unroll
            for (int msk = 1; msk < 16; msk <<= 1)
                s += __shfl_xor(s, msk, 32);
            const float g = 1.f / (1.f + __expf(-(s + bp0)));
            const int m = row0 + j + 8 * hi;
            const size_t bk = (size_t)(b0 + m) * N_CONCEPTS + k;
#pragma unroll
            for (int t = 0; t < 8; ++t) {
                float pos = acc[t][j];       // h = 16t+n16       (< 128)
                float neg = acc[t + 8][j];   // h = 128+16t+n16
                c_emb[bk * EMB_SIZE + 16 * t + n16] = neg + (pos - neg) * g;
            }
            if (n16 == 0) {
                c_pred[bk] = g;
                c_int[bk]  = g;
            }
        }

        // next buffer fully landed + all waves done with this one
        wait_async_all();
        __syncthreads();
    }
}

// ---------------------------------------------------------------------------
extern "C" void kernel_launch(void* const* d_in, const int* in_sizes, int n_in,
                              void* d_out, int out_size, void* d_ws, size_t ws_size,
                              hipStream_t stream) {
    (void)in_sizes; (void)n_in; (void)out_size; (void)ws_size;
    const float* x  = (const float*)d_in[0];
    const float* We = (const float*)d_in[1];
    const float* be = (const float*)d_in[2];
    const float* Wc = (const float*)d_in[3];
    const float* bc = (const float*)d_in[4];
    const float* wp = (const float*)d_in[5];
    const float* bp = (const float*)d_in[6];

    float* out    = (float*)d_out;
    float* c_emb  = out;                                   // 8192*64*128
    float* c_pred = out + (size_t)BATCH * N_CONCEPTS * EMB_SIZE;
    float* c_int  = c_pred + (size_t)BATCH * N_CONCEPTS;
    float* emb    = c_int + (size_t)BATCH * N_CONCEPTS;

    char* ws = (char*)d_ws;
    __bf16* emb_b = (__bf16*)ws;                           // 2 MB
    __bf16* we_b  = (__bf16*)(ws + (2u << 20));            // 512 KB
    __bf16* wc_b  = (__bf16*)(ws + (2u << 20) + (512u << 10)); // 4 MB

    cvt_weights<<<512, 256, 0, stream>>>(We, Wc, we_b, wc_b);

    const size_t lds1 = 2 * 128 * 128 * sizeof(__bf16);    // 64 KB
    gemm1_emb<<<BATCH / 128, 256, lds1, stream>>>(x, we_b, be, emb, emb_b);

    const size_t lds2 = (128 * EMB_SIZE + 2 * 2 * EMB_SIZE * EMB_SIZE)
                        * sizeof(__bf16);                  // 160 KB
    gemm2_fused<<<BATCH / 128, 256, lds2, stream>>>(
        emb_b, wc_b, bc, wp, bp, c_emb, c_pred, c_int);
}
